// CouplingLayer_86552180949780
// MI455X (gfx1250) — matrix operand order repaired and checked by reading, not compiled, not run on hardware
//
#include <hip/hip_runtime.h>
#include <hip/hip_bf16.h>

// ---------------------------------------------------------------------------
// CDNA5 (gfx1250) fused RQS coupling layer.
//  - bf16 WMMA (v_wmma_f32_16x16x32_bf16), wave32 fragments, contiguous k order (k = 16*(lane/16) + i) in A and B
//  - B tiles stored in LDS pre-swizzled to fragment order -> ds_load_b128
//  - GEMM2 A tile (bf16 h-matrix, 2D strided tile) fetched by the Tensor Data
//    Mover (tensor_load_to_lds + s_wait_tensorcnt)
//  - GEMM2 epilogue fuses softmax/softplus + rational-quadratic spline to
//    avoid materializing the 100MB parameter tensor (saves ~400MB HBM traffic)
// ---------------------------------------------------------------------------

typedef __attribute__((ext_vector_type(16))) __bf16    v16bf;
typedef __attribute__((ext_vector_type(8)))  float     v8f;
typedef __attribute__((ext_vector_type(4)))  unsigned  su4;
typedef __attribute__((ext_vector_type(8)))  unsigned  su8;

#define BT 256
#define NF 4096
#define HID 1024
#define C3K 98304   // NF * 3 * K_BINS
#define BND 8.0f

__device__ __forceinline__ unsigned short f2bf_us(float f) {
  unsigned int u = __float_as_uint(f);
  unsigned int r = (u + 0x7FFFu + ((u >> 16) & 1u)) >> 16;
  return (unsigned short)r;
}
__device__ __forceinline__ __bf16 us2bf(unsigned short s) {
  union { unsigned short u; __bf16 b; } v; v.u = s; return v.b;
}
__device__ __forceinline__ __bf16 f2bf(float f) { return us2bf(f2bf_us(f)); }

// ---------------------------------------------------------------------------
// ld_acc[b] = log_det_in[b]
__global__ void init_ld_kernel(const float* __restrict__ ld_in,
                               float* __restrict__ ld_acc) {
  int i = threadIdx.x;
  if (i < BT) ld_acc[i] = ld_in[i];
}

// xin_bf16[i] = bf16(z[i] * mask[i % NF])
__global__ void prep_in_kernel(const float* __restrict__ z,
                               const int* __restrict__ mask,
                               unsigned short* __restrict__ xin) {
  int i = blockIdx.x * blockDim.x + threadIdx.x;
  if (i < BT * NF) {
    float v = z[i] * (float)mask[i & (NF - 1)];
    xin[i] = f2bf_us(v);
  }
}

// ---------------------------------------------------------------------------
// GEMM1: H = relu(X @ W1 + b1), X bf16 [256,4096], W1 f32 [4096,1024]
// WG tile 64x64, 8 waves, each wave: 16M x 32N (2 accumulators)
__global__ __launch_bounds__(256) void gemm1_relu_kernel(
    const unsigned short* __restrict__ X,
    const float* __restrict__ W1,
    const float* __restrict__ b1,
    unsigned short* __restrict__ Hout) {
  __shared__ __bf16 As[64][32];
  __shared__ __bf16 Bs[4][32][16];   // [nt][lane = n | khalf<<4][k & 15]
  const int t = threadIdx.x;
  const int lane = t & 31, w = t >> 5;
  const int bm = blockIdx.x >> 4;   // 0..3
  const int bn = blockIdx.x & 15;   // 0..15
  const int row0 = bm * 64, col0 = bn * 64;
  const int m_off = (w & 3) * 16;
  const int n_off = (w >> 2) * 32;
  v8f acc[2] = {};

  for (int kk = 0; kk < NF; kk += 32) {
#pragma unroll
    for (int i = 0; i < 8; ++i) {           // A tile 64x32 bf16
      int e = t * 8 + i;
      int r = e >> 5, c = e & 31;
      As[r][c] = us2bf(X[(row0 + r) * NF + kk + c]);
    }
#pragma unroll
    for (int i = 0; i < 8; ++i) {           // B tile 32x64, f32->bf16, swizzled
      int e = t * 8 + i;
      int k = e >> 6, j = e & 63;
      Bs[j >> 4][(j & 15) | ((k >> 4) << 4)][k & 15] =
          f2bf(W1[(size_t)(kk + k) * HID + col0 + j]);
    }
    __syncthreads();

    // A fragment: lane = M row, K half selected by lane>>4
    const int m = m_off + (lane & 15);
    const int kbA = (lane >> 4) * 16;
    v16bf af;
#pragma unroll
    for (int i = 0; i < 16; ++i)
      af[i] = As[m][kbA + i];

#pragma unroll
    for (int nt = 0; nt < 2; ++nt) {
      const int ntg = (w >> 2) * 2 + nt;
      v16bf bfv = *(const v16bf*)&Bs[ntg][lane][0];
      acc[nt] = __builtin_amdgcn_wmma_f32_16x16x32_bf16(
          false, af, false, bfv, (short)0, acc[nt], false, false);
    }
    __syncthreads();
  }

#pragma unroll
  for (int nt = 0; nt < 2; ++nt) {
#pragma unroll
    for (int r = 0; r < 8; ++r) {
      int rl = m_off + r + 8 * (lane >> 4);
      int cl = n_off + nt * 16 + (lane & 15);
      float v = acc[nt][r] + b1[col0 + cl];
      v = v > 0.f ? v : 0.f;
      Hout[(row0 + rl) * HID + col0 + cl] = f2bf_us(v);
    }
  }
}

// ---------------------------------------------------------------------------
// GEMM2 + fused spline epilogue.
// C tile: 256 (all rows) x 96 (4 spline elements x 24 params).
// 8 waves; wave w covers rows 32w..32w+31 => 2 Mtiles x 6 Ntiles = 12 accums.
// A tile (256x32 bf16, row stride 1024) is loaded by the Tensor Data Mover.
__global__ __launch_bounds__(256) void gemm2_spline_kernel(
    const unsigned short* __restrict__ Hin,   // [256,1024] bf16
    const float* __restrict__ W2,             // [1024,98304]
    const float* __restrict__ b2,             // [98304]
    const float* __restrict__ z,              // [256,4096]
    const int*   __restrict__ mask,           // [4096]
    int pass,                                 // 0 or 1
    float* __restrict__ Yout,                 // [256,4096] spline output
    unsigned short* __restrict__ Ybf,         // bf16 copy (pass0) or nullptr
    float* __restrict__ ldacc) {              // [256] log-det accumulator
  struct Tiles {
    __bf16 A[BT][32];
    __bf16 B[6][32][16];   // [nt][lane = n | khalf<<4][k & 15]
  };
  __shared__ union SM { Tiles tl; float C[BT][96]; } sm;

  const int t = threadIdx.x;
  const int lane = t & 31, w = t >> 5;
  const int col0 = blockIdx.x * 96;
  v8f acc[2][6] = {};

  for (int kk = 0; kk < HID; kk += 32) {
    // --- TDM: async copy A tile (2D: 32 elem/row x 256 rows, stride 1024) ---
    if (t < 32) {
      unsigned ldsA = (unsigned)(unsigned long long)(&sm.tl.A[0][0]);
      unsigned long long ga = (unsigned long long)(const void*)(Hin + kk);
      su4 g0;
      g0[0] = 1u;                                            // count=1
      g0[1] = ldsA;                                          // lds_addr
      g0[2] = (unsigned)(ga & 0xFFFFFFFFull);                // global_addr lo
      g0[3] = (unsigned)((ga >> 32) & 0x1FFFFFFull)          // global_addr hi
              | (2u << 30);                                  // type=2 (image)
      su8 g1;
      g1[0] = (1u << 16);        // workgroup_mask=0, data_size=1 (2 bytes)
      g1[1] = (1024u << 16);     // tensor_dim0 = 1024 (bits 79:48, low part)
      g1[2] = (256u << 16);      // tensor_dim0 hi=0 ; tensor_dim1 = 256
      g1[3] = (32u << 16);       // tensor_dim1 hi=0 ; tile_dim0 = 32
      g1[4] = 256u;              // tile_dim1 = 256 ; tile_dim2 = 0
      g1[5] = 1024u;             // tensor_dim0_stride = 1024 (low 32)
      g1[6] = 0u;                // stride hi ; tensor_dim1_stride low
      g1[7] = 0u;
      asm volatile("tensor_load_to_lds %0, %1" :: "s"(g0), "s"(g1) : "memory");
    }

    // --- B tile 32x96, f32 -> bf16, stored swizzled to fragment order ---
#pragma unroll
    for (int i = 0; i < 12; ++i) {
      int e = t * 12 + i;
      int k = e / 96, j = e - k * 96;
      sm.tl.B[j >> 4][(j & 15) | ((k >> 4) << 4)][k & 15] =
          f2bf(W2[(size_t)(kk + k) * C3K + col0 + j]);
    }

    if (t < 32) __builtin_amdgcn_s_wait_tensorcnt(0);
    __syncthreads();

    v16bf bfv[6];
#pragma unroll
    for (int nt = 0; nt < 6; ++nt)
      bfv[nt] = *(const v16bf*)&sm.tl.B[nt][lane][0];

    const int kbA = (lane >> 4) * 16;
#pragma unroll
    for (int mt = 0; mt < 2; ++mt) {
      const int m = w * 32 + mt * 16 + (lane & 15);
      v16bf af;
#pragma unroll
      for (int i = 0; i < 16; ++i)
        af[i] = sm.tl.A[m][kbA + i];
#pragma unroll
      for (int nt = 0; nt < 6; ++nt)
        acc[mt][nt] = __builtin_amdgcn_wmma_f32_16x16x32_bf16(
            false, af, false, bfv[nt], (short)0, acc[mt][nt], false, false);
    }
    __syncthreads();
  }

  // Stage C tile (overwrites the A/B tiles; barriers above/below protect it)
#pragma unroll
  for (int mt = 0; mt < 2; ++mt)
#pragma unroll
    for (int nt = 0; nt < 6; ++nt)
#pragma unroll
      for (int r = 0; r < 8; ++r) {
        int row = w * 32 + mt * 16 + r + 8 * (lane >> 4);
        int col = nt * 16 + (lane & 15);
        sm.C[row][col] = acc[mt][nt][r];
      }
  __syncthreads();

  // Spline epilogue: thread t owns batch row t, 4 spline elements
  float ldsum = 0.f;
  for (int j = 0; j < 4; ++j) {
    const int n = blockIdx.x * 4 + j;
    float o[24];
#pragma unroll
    for (int i = 0; i < 24; ++i)
      o[i] = sm.C[t][j * 24 + i] + b2[col0 + j * 24 + i];

    // softmax(o[0:8]) * 2B, softmax(o[8:16]) * 2B, softplus(o[16:24])
    float mw = o[0], mh = o[8];
#pragma unroll
    for (int i = 1; i < 8; ++i) { mw = fmaxf(mw, o[i]); mh = fmaxf(mh, o[8 + i]); }
    float Wp[8], Hp[8], Dp[8];
    float sw = 0.f, sh = 0.f;
#pragma unroll
    for (int i = 0; i < 8; ++i) {
      Wp[i] = __expf(o[i] - mw);     sw += Wp[i];
      Hp[i] = __expf(o[8 + i] - mh); sh += Hp[i];
      float s = o[16 + i];
      Dp[i] = fmaxf(s, 0.f) + log1pf(__expf(-fabsf(s)));   // softplus
    }
    float rw = (2.0f * BND) / sw, rh = (2.0f * BND) / sh;
#pragma unroll
    for (int i = 0; i < 8; ++i) { Wp[i] *= rw; Hp[i] *= rh; }

    float mul = (float)(mask[n] + (pass == 0 ? 1 : 0));
    float xraw = z[t * NF + n] * mul;
    bool inside = (xraw >= -BND) && (xraw <= BND);
    float xs = fminf(fmaxf(xraw, -BND), BND);

    // bin search (idx = last k with xs >= left knot)
    float cx = -BND, cy = -BND;
    float xk = -BND, yk = -BND, wk = Wp[0], hk = Hp[0], dk = Dp[0], dk1 = Dp[1];
#pragma unroll
    for (int k = 0; k < 8; ++k) {
      if (xs >= cx) {
        xk = cx; yk = cy; wk = Wp[k]; hk = Hp[k];
        dk = Dp[k]; dk1 = (k < 7) ? Dp[k + 1] : 1.0f;
      }
      cx += Wp[k]; cy += Hp[k];
    }

    float sk  = hk / wk;
    float th  = (xs - xk) / wk;
    float tt  = th * (1.f - th);
    float den = sk + (dk1 + dk - 2.f * sk) * tt;
    float y   = yk + hk * (sk * th * th + dk * tt) / den;
    float omt = 1.f - th;
    float num = sk * sk * (dk1 * th * th + 2.f * sk * tt + dk * omt * omt);
    float ld  = __logf(num) - 2.f * __logf(den);

    float yfin = inside ? y : xraw;
    Yout[t * NF + n] = yfin;
    if (Ybf) Ybf[t * NF + n] = f2bf_us(yfin);
    ldsum += inside ? ld : 0.f;
  }
  atomicAdd(&ldacc[t], ldsum);
}

// ---------------------------------------------------------------------------
// out[0 : BT*NF] = z_in2 + z_out2 ; out[BT*NF : +BT] = log_det
__global__ void final_combine_kernel(const float* __restrict__ zin2,
                                     const float* __restrict__ zout2,
                                     const float* __restrict__ ldacc,
                                     float* __restrict__ out) {
  int i = blockIdx.x * blockDim.x + threadIdx.x;
  if (i < BT * NF) out[i] = zin2[i] + zout2[i];
  if (i < BT) out[BT * NF + i] = ldacc[i];
}

// ---------------------------------------------------------------------------
extern "C" void kernel_launch(void* const* d_in, const int* in_sizes, int n_in,
                              void* d_out, int out_size, void* d_ws, size_t ws_size,
                              hipStream_t stream) {
  const float* z    = (const float*)d_in[0];
  const float* ld0  = (const float*)d_in[1];
  const float* w1   = (const float*)d_in[2];
  const float* b1   = (const float*)d_in[3];
  const float* w2   = (const float*)d_in[4];
  const float* b2   = (const float*)d_in[5];
  const int*   mask = (const int*)d_in[6];
  float* out = (float*)d_out;

  char* ws = (char*)d_ws;
  unsigned short* xin_bf  = (unsigned short*)(ws);                         // 2 MB
  unsigned short* zo2_bf  = (unsigned short*)(ws + (((size_t)2) << 20));   // 2 MB
  unsigned short* h_bf    = (unsigned short*)(ws + (((size_t)4) << 20));   // 512 KB
  float* zout2 = (float*)(ws + (((size_t)4) << 20) + (((size_t)512) << 10)); // 4 MB
  float* zin2  = zout2 + (size_t)BT * NF;                                  // 4 MB
  float* ldacc = zin2 + (size_t)BT * NF;                                   // 1 KB

  init_ld_kernel<<<1, 256, 0, stream>>>(ld0, ldacc);
  prep_in_kernel<<<(BT * NF) / 256, 256, 0, stream>>>(z, mask, xin_bf);

  // ---- pass 0: params from net(z*m), spline applied to z*(m+1) ----
  gemm1_relu_kernel<<<64, 256, 0, stream>>>(xin_bf, w1, b1, h_bf);
  gemm2_spline_kernel<<<C3K / 96, 256, 0, stream>>>(
      h_bf, w2, b2, z, mask, /*pass=*/0, zout2, zo2_bf, ldacc);

  // ---- pass 1: params from net(z_out2), spline applied to z*m ----
  gemm1_relu_kernel<<<64, 256, 0, stream>>>(zo2_bf, w1, b1, h_bf);
  gemm2_spline_kernel<<<C3K / 96, 256, 0, stream>>>(
      h_bf, w2, b2, z, mask, /*pass=*/1, zin2, (unsigned short*)nullptr, ldacc);

  final_combine_kernel<<<(BT * NF) / 256, 256, 0, stream>>>(zin2, zout2, ldacc, out);
}
